// MoELayer_50130858279275
// MI455X (gfx1250) — compile-verified, matrix operand order checked by reading, not run on hardware
//
#include <hip/hip_runtime.h>

typedef __attribute__((ext_vector_type(16))) __bf16 v16bf;
typedef __attribute__((ext_vector_type(8)))  __bf16 v8bf;
typedef __attribute__((ext_vector_type(8)))  float  v8f;
typedef int v4i __attribute__((vector_size(16)));   // matches builtin's int4

constexpr int Bsz = 2, Ssz = 2048, Dm = 1024, Ne = 8, Ii = 2816;
constexpr int Tt = Bsz * Ssz;      // 4096 tokens
constexpr int TM = 32;             // token tile rows (2 row-tiles of 16)
constexpr int KI = 128;            // I-dimension chunk per pipeline step
constexpr int TB = 512;            // threads per FFN block = 16 waves

// ---------------------------------------------------------------------------
// Gating: one wave (32 lanes) per token.
// ---------------------------------------------------------------------------
__global__ __launch_bounds__(256) void moe_gate(const float* __restrict__ x,
                                                const float* __restrict__ gw,
                                                float* __restrict__ combine,
                                                float* __restrict__ gsum,
                                                float* __restrict__ cnt)
{
    const int lane = threadIdx.x & 31;
    const int wave = threadIdx.x >> 5;
    const int t = blockIdx.x * 8 + wave;
    if (t >= Tt) return;

    float s[Ne];
#pragma unroll
    for (int e = 0; e < Ne; ++e) s[e] = 0.f;

    const float* xr = x + (size_t)t * Dm;
    for (int d = lane; d < Dm; d += 32) {
        float xv = xr[d];
#pragma unroll
        for (int e = 0; e < Ne; ++e) s[e] += xv * gw[e * Dm + d];
    }
#pragma unroll
    for (int e = 0; e < Ne; ++e) {
#pragma unroll
        for (int off = 16; off > 0; off >>= 1) s[e] += __shfl_xor(s[e], off, 32);
    }

    if (lane == 0) {
        float m = s[0];
#pragma unroll
        for (int e = 1; e < Ne; ++e) m = fmaxf(m, s[e]);
        float g[Ne]; float den = 0.f;
#pragma unroll
        for (int e = 0; e < Ne; ++e) { g[e] = __expf(s[e] - m); den += g[e]; }
        const float inv = __builtin_amdgcn_rcpf(den);

        int i0 = 0;
#pragma unroll
        for (int e = 1; e < Ne; ++e) if (s[e] > s[i0]) i0 = e;
        int i1 = (i0 == 0) ? 1 : 0;
#pragma unroll
        for (int e = 0; e < Ne; ++e) if (e != i0 && s[e] > s[i1]) i1 = e;

        const float a   = __expf(s[i1] - s[i0]);
        const float rs  = __builtin_amdgcn_rcpf(1.f + a);
        const float w0  = rs;
        const float w1v = a * rs;

#pragma unroll
        for (int e = 0; e < Ne; ++e) {
            float c = (e == i0) ? w0 : ((e == i1) ? w1v : 0.f);
            combine[t * Ne + e] = c;
            atomicAdd(&gsum[e], g[e] * inv);
        }
        atomicAdd(&cnt[i0], 1.f);
        atomicAdd(&cnt[i1], 1.f);
    }
}

__global__ void moe_init(float* p) { if (threadIdx.x < 16) p[threadIdx.x] = 0.f; }

__global__ void moe_loss(const float* __restrict__ gsum,
                         const float* __restrict__ cnt,
                         float* __restrict__ out)
{
    float acc = 0.f;
    for (int e = 0; e < Ne; ++e)
        acc += (cnt[e] / (float)Tt) * (gsum[e] / (float)Tt);
    out[0] = (float)Ne * acc;
}

// ---------------------------------------------------------------------------
// fp32 -> bf16 bulk conversion (weights + activations into workspace).
// bf16 weights (138 MB) fit in the 192 MB L2; fp32 (277 MB) does not.
// ---------------------------------------------------------------------------
__global__ __launch_bounds__(256) void moe_cvt(const float* __restrict__ src,
                                               __bf16* __restrict__ dst, size_t n)
{
    size_t i = ((size_t)blockIdx.x * blockDim.x + threadIdx.x) * 8;
    if (i + 8 <= n) {
        float4 a = *(const float4*)(src + i);
        float4 b = *(const float4*)(src + i + 4);
        v8bf o;
        o[0] = (__bf16)a.x; o[1] = (__bf16)a.y; o[2] = (__bf16)a.z; o[3] = (__bf16)a.w;
        o[4] = (__bf16)b.x; o[5] = (__bf16)b.y; o[6] = (__bf16)b.z; o[7] = (__bf16)b.w;
        *(v8bf*)(dst + i) = o;
    }
}

// ---------------------------------------------------------------------------
template <typename WT>
__device__ __forceinline__ v16bf load_b(const WT* p)
{
    if constexpr (sizeof(WT) == 2) {
        return *(const v16bf*)p;
    } else {
        v16bf b;
#pragma unroll
        for (int j = 0; j < 16; ++j) b[j] = (__bf16)p[j];
        return b;
    }
}

// ---------------------------------------------------------------------------
// Expert FFN. One block per 32-token tile, 16 waves.
//   stage1: wave w -> I-chunk (w&7), row-tile (w>>3); h tile 16x16 -> LDS.
//   stage2: wave w owns 64 D-columns x 32 rows (8 acc tiles, 64 VGPRs);
//           each w3 B-tile feeds 2 WMMAs (both row tiles).
// ---------------------------------------------------------------------------
template <typename WT>
__global__ __launch_bounds__(TB, 1) void moe_ffn(const WT* __restrict__ xq,
                                                 const WT* __restrict__ w1,
                                                 const WT* __restrict__ w2,
                                                 const WT* __restrict__ w3,
                                                 const float* __restrict__ combine,
                                                 float* __restrict__ out)
{
    __shared__ __align__(32) __bf16 xs[TM][Dm];   // 64 KB token tile (bf16)
    __shared__ __align__(32) __bf16 hs[TM][KI];   //  8 KB h tile (bf16)
    __shared__ float cs[TM];

    const int tid    = threadIdx.x;
    const int lane   = tid & 31;
    const int wave   = tid >> 5;         // 0..15
    const int tok0   = blockIdx.x * TM;
    const int n      = lane & 15;
    const int khalf  = (lane >> 4) << 4; // lanes 16..31 take the upper K half
    const int rowoff = (lane >> 4) << 3; // C/D layout: VGPR r -> rows r, r+8
    const int ichunk = wave & 7;         // stage1: which 16 I-columns
    const int m0     = (wave >> 3) << 4; // stage1: row-tile base (0 or 16)
    const int dbase  = wave * 64;        // stage2: this wave's 64 D-columns

    // ---- stage token tile into LDS ----
    if constexpr (sizeof(WT) == 2) {
        const __bf16* gx = xq + (size_t)tok0 * Dm;  // 64 KB contiguous
#if __has_builtin(__builtin_amdgcn_global_load_async_to_lds_b128)
        for (int i = tid; i < TM * Dm / 8; i += TB) {
            __builtin_amdgcn_global_load_async_to_lds_b128(
                (__attribute__((address_space(1))) v4i*)(gx + (size_t)i * 8),
                (__attribute__((address_space(3))) v4i*)(&xs[0][0] + (size_t)i * 8),
                0, 0);
        }
        asm volatile("s_wait_asynccnt 0x0" ::: "memory");
#else
        for (int i = tid; i < TM * Dm / 8; i += TB)
            *(v8bf*)(&xs[0][0] + (size_t)i * 8) = *(const v8bf*)(gx + (size_t)i * 8);
#endif
    } else {
        for (int i = tid; i < TM * Dm; i += TB) {
            int r = i >> 10, c = i & (Dm - 1);
            xs[r][c] = (__bf16)xq[(size_t)(tok0 + r) * Dm + c];
        }
    }
    __syncthreads();

    v8f acc[2][4];                       // [row-tile][col-tile], 64 VGPRs
    v8f vzero = {};
#pragma unroll
    for (int rt = 0; rt < 2; ++rt)
#pragma unroll
        for (int j = 0; j < 4; ++j) acc[rt][j] = vzero;

    for (int e = 0; e < Ne; ++e) {
        if (tid < TM) cs[tid] = combine[(tok0 + tid) * Ne + e];
        __syncthreads();
        float crow[8];
#pragma unroll
        for (int r = 0; r < 8; ++r) crow[r] = cs[m0 + r + rowoff];

        const WT* w1e = w1 + (size_t)e * Ii * Dm;
        const WT* w2e = w2 + (size_t)e * Ii * Dm;
        const WT* w3e = w3 + (size_t)e * Dm * Ii;

        for (int kk = 0; kk < Ii; kk += KI) {
            const int iw = kk + ichunk * 16;
            v8f hacc = vzero, gacc = vzero;
            const WT* p1 = w1e + (size_t)(iw + n) * Dm + khalf;
            const WT* p2 = w2e + (size_t)(iw + n) * Dm + khalf;

            for (int d = 0; d < Dm; d += 32) {
                const v16bf a  = *(const v16bf*)&xs[m0 + n][d + khalf];
                const v16bf b1 = load_b(p1 + d);
                const v16bf b2 = load_b(p2 + d);
                __builtin_prefetch(p1 + d + 512, 0, 3);   // global_prefetch_b8 (WGP)
                __builtin_prefetch(p2 + d + 512, 0, 3);
                hacc = __builtin_amdgcn_wmma_f32_16x16x32_bf16(
                           false, a, false, b1, (short)0, hacc, false, false);
                gacc = __builtin_amdgcn_wmma_f32_16x16x32_bf16(
                           false, a, false, b2, (short)0, gacc, false, false);
            }

            // h = combine * silu(x.w1) * (x.w2); fast rcp path.
#pragma unroll
            for (int r = 0; r < 8; ++r) {
                float hv = hacc[r];
                float sv = hv * __builtin_amdgcn_rcpf(1.f + __expf(-hv));
                hs[m0 + r + rowoff][ichunk * 16 + n] = (__bf16)(crow[r] * sv * gacc[r]);
            }
            __syncthreads();

            // stage2: acc[32 x 64] += h[32,128] . w3[cols,128]^T
#pragma unroll
            for (int kc = 0; kc < KI; kc += 32) {
                const v16bf a0 = *(const v16bf*)&hs[n][kc + khalf];
                const v16bf a1 = *(const v16bf*)&hs[16 + n][kc + khalf];
#pragma unroll
                for (int j = 0; j < 4; ++j) {
                    const WT* p3 = w3e + (size_t)(dbase + j * 16 + n) * Ii + kk + kc + khalf;
                    const v16bf b3 = load_b(p3);
                    acc[0][j] = __builtin_amdgcn_wmma_f32_16x16x32_bf16(
                                    false, a0, false, b3, (short)0, acc[0][j], false, false);
                    acc[1][j] = __builtin_amdgcn_wmma_f32_16x16x32_bf16(
                                    false, a1, false, b3, (short)0, acc[1][j], false, false);
                }
            }
            __syncthreads();
        }
    }

#pragma unroll
    for (int rt = 0; rt < 2; ++rt)
#pragma unroll
        for (int j = 0; j < 4; ++j)
#pragma unroll
            for (int r = 0; r < 8; ++r)
                out[(size_t)(tok0 + rt * 16 + r + rowoff) * Dm + dbase + j * 16 + n]
                    = acc[rt][j][r];
}

// ---------------------------------------------------------------------------
extern "C" void kernel_launch(void* const* d_in, const int* in_sizes, int n_in,
                              void* d_out, int out_size, void* d_ws, size_t ws_size,
                              hipStream_t stream)
{
    const float* x  = (const float*)d_in[0];
    const float* gw = (const float*)d_in[1];
    const float* w1 = (const float*)d_in[2];
    const float* w2 = (const float*)d_in[3];
    const float* w3 = (const float*)d_in[4];
    float* out = (float*)d_out;

    const size_t nW12 = (size_t)Ne * Ii * Dm;   // 23,068,672 elements
    const size_t nW3  = (size_t)Ne * Dm * Ii;
    const size_t nX   = (size_t)Tt * Dm;

    char* ws = (char*)d_ws;
    size_t off = 0;
    float* combine = (float*)(ws + off); off += (size_t)Tt * Ne * 4;   // 128 KB
    float* gsum    = (float*)(ws + off); off += 128;                   // stats
    __bf16* xb  = (__bf16*)(ws + off); off += nX  * 2;
    __bf16* w1b = (__bf16*)(ws + off); off += nW12 * 2;
    __bf16* w2b = (__bf16*)(ws + off); off += nW12 * 2;
    __bf16* w3b = (__bf16*)(ws + off); off += nW3  * 2;
    const bool preconv = (ws_size >= off);       // ~146.6 MB needed

    moe_init<<<1, 32, 0, stream>>>(gsum);
    moe_gate<<<Tt / 8, 256, 0, stream>>>(x, gw, combine, gsum, gsum + Ne);

    if (preconv) {
        moe_cvt<<<(int)(nX   / 2048), 256, 0, stream>>>(x,  xb,  nX);
        moe_cvt<<<(int)(nW12 / 2048), 256, 0, stream>>>(w1, w1b, nW12);
        moe_cvt<<<(int)(nW12 / 2048), 256, 0, stream>>>(w2, w2b, nW12);
        moe_cvt<<<(int)(nW3  / 2048), 256, 0, stream>>>(w3, w3b, nW3);
        moe_ffn<__bf16><<<Tt / TM, TB, 0, stream>>>(xb, w1b, w2b, w3b, combine, out);
    } else {
        moe_ffn<float><<<Tt / TM, TB, 0, stream>>>(x, w1, w2, w3, combine, out);
    }

    moe_loss<<<1, 1, 0, stream>>>(gsum, gsum + Ne, out + (size_t)Tt * Dm);
}
